// EnsemblesWithMessagePassing_2104533975419
// MI455X (gfx1250) — matrix operand order repaired and checked by reading, not compiled, not run on hardware
//
#include <hip/hip_runtime.h>
#include <hip/hip_bf16.h>

// ---------------------------------------------------------------------------
// EnsemblesWithMessagePassing on MI455X (gfx1250, wave32, WMMA + async-LDS)
//
//   0. convert tokens, w_net -> bf16; transpose-convert wq/wkv/wout -> [n][k]
//   1. tn   = rmsnorm(tokens)                          (VALU)
//   2. onet = tokens @ w_net^T + b_net per member      (WMMA, async staging)
//   3. q    = tn @ wq                                  (WMMA)
//   4. kv[0:8] = tokens @ wkv ; kv[8:16] = onet @ wkv  (WMMA)
//   5. per-head rmsnorm of k-half                      (VALU)
//   6. softmax attention (M=16) + sigmoid gate         (VALU)
//   7. pooled = (o*gate) @ wout -> d_out               (WMMA)
//
// GEMM: block tile 128x64, 8 waves as 4x2, wave tile 32x32 (4 accums),
// K-stage 64 (2 WMMA K-chunks), double-buffered global_load_async_to_lds_b64,
// s_wait_asynccnt pipelining -> 8 v_wmma per barrier pair per wave.
// ---------------------------------------------------------------------------

typedef __bf16 bf16_t;
typedef __attribute__((ext_vector_type(16))) __bf16 v16bf;
typedef __attribute__((ext_vector_type(8)))  float  v8f;

#define L_     8
#define B_     2
#define N_     512
#define D_     1024
#define H_     8
#define DH_    64
#define INNER_ 512
#define M_     16
#define EPS_   1.1920928955078125e-07f

#define LDSK   68     // bf16 row stride: 136B rows, 8B aligned chunks

// ---- async global->LDS copy (8 bytes per lane), tracked by ASYNCcnt -------
__device__ __forceinline__ void async_b64(unsigned lds_off, const bf16_t* g) {
    asm volatile("global_load_async_to_lds_b64 %0, %1, off"
                 :: "v"(lds_off), "v"(g) : "memory");
}

#if defined(__has_builtin)
#if __has_builtin(__builtin_amdgcn_s_wait_asynccnt)
#define WAIT_ASYNC(n) __builtin_amdgcn_s_wait_asynccnt(n)
#endif
#endif
#ifndef WAIT_ASYNC
#define WAIT_ASYNC(n) asm volatile("s_wait_asynccnt %0" :: "i"(n) : "memory")
#endif

// ---------------------------------------------------------------------------
// bf16 WMMA GEMM, double-buffered async-LDS staging.
//   A  : bf16 [rows][K]   (lda == Kdim)
//   Bt : bf16 [n][K]      (B transposed, K contiguous)
//   C  : CT   [rows][N]   (ldc == Ndim)
// rows % 128 == 0, Ndim % 64 == 0, Kdim % 64 == 0.
// ---------------------------------------------------------------------------
template<bool HASBIAS, typename CT>
__global__ __launch_bounds__(256)
void gemm_wmma(const bf16_t* __restrict__ A, const bf16_t* __restrict__ Bt,
               const float* __restrict__ bias, CT* __restrict__ C,
               int Ndim, int Kdim,
               long aStrideZ, long bStrideZ, long biasStrideZ, long cStrideZ)
{
    __shared__ bf16_t Alds[2][128][LDSK];
    __shared__ bf16_t Blds[2][64][LDSK];

    const int z = blockIdx.z;
    A  += (long)z * aStrideZ;
    Bt += (long)z * bStrideZ;
    C  += (long)z * cStrideZ;

    const int tid    = threadIdx.x;
    const int lane   = tid & 31;
    const int wave   = tid >> 5;
    const int rowBlk = blockIdx.y * 128;
    const int colBlk = blockIdx.x * 64;
    const int wm     = wave & 3;       // 4 row groups of 32
    const int wn     = wave >> 2;      // 2 col groups of 32
    const int half   = lane >> 4;
    const int l16    = lane & 15;

    // ---- async staging: A 128x64 (8 b64/thread), B 64x64 (4 b64/thread) ---
    auto issue = [&](int b, int ko) {
        #pragma unroll
        for (int i = 0; i < 8; ++i) {
            int c = tid + i * 256;             // chunk id, 16 chunks per row
            int r = c >> 4, k4 = (c & 15) * 4;
            async_b64((unsigned)(size_t)&Alds[b][r][k4],
                      A + (long)(rowBlk + r) * Kdim + ko + k4);
        }
        #pragma unroll
        for (int i = 0; i < 4; ++i) {
            int c = tid + i * 256;
            int r = c >> 4, k4 = (c & 15) * 4;
            async_b64((unsigned)(size_t)&Blds[b][r][k4],
                      Bt + (long)(colBlk + r) * Kdim + ko + k4);
        }
    };

    v8f acc00 = {}, acc01 = {}, acc10 = {}, acc11 = {};

    // ---- 8 WMMAs per staged tile ----
    auto compute = [&](int cur) {
        #pragma unroll
        for (int ks = 0; ks < 2; ++ks) {       // two K=32 chunks
            const int ko = ks * 32;
            union { v16bf v; unsigned u[8]; } a0, a1, b0, b1;
            const int ar0 = wm * 32 + l16;
            const int c0  = wn * 32 + l16;
            #pragma unroll
            for (int p = 0; p < 8; ++p) {      // A frag: VGPR p = K kb,kb+1
                int kb = ko + ((p >> 2) * 16) + half * 8 + ((p & 3) * 2);
                a0.u[p] = *(const unsigned*)&Alds[cur][ar0][kb];
                a1.u[p] = *(const unsigned*)&Alds[cur][ar0 + 16][kb];
            }
            #pragma unroll
            for (int p = 0; p < 8; ++p) {      // B frag: VGPR p = K kb,kb+1
                int kb = ko + ((p >> 2) * 16) + half * 8 + ((p & 3) * 2);
                b0.u[p] = *(const unsigned*)&Blds[cur][c0][kb];
                b1.u[p] = *(const unsigned*)&Blds[cur][c0 + 16][kb];
            }
            acc00 = __builtin_amdgcn_wmma_f32_16x16x32_bf16(
                        false, a0.v, false, b0.v, (short)0, acc00, false, false);
            acc01 = __builtin_amdgcn_wmma_f32_16x16x32_bf16(
                        false, a0.v, false, b1.v, (short)0, acc01, false, false);
            acc10 = __builtin_amdgcn_wmma_f32_16x16x32_bf16(
                        false, a1.v, false, b0.v, (short)0, acc10, false, false);
            acc11 = __builtin_amdgcn_wmma_f32_16x16x32_bf16(
                        false, a1.v, false, b1.v, (short)0, acc11, false, false);
        }
    };

    const int nt = Kdim >> 6;                  // K-stages of 64
    issue(0, 0);                               // prologue
    for (int i = 0; i < nt - 1; ++i) {
        issue((i + 1) & 1, (i + 1) << 6);      // next tile into other buffer
        WAIT_ASYNC(12);                        // tile i's 12 copies landed
        __syncthreads();
        compute(i & 1);
        __syncthreads();                       // buf reusable next iteration
    }
    WAIT_ASYNC(0);                             // last tile
    __syncthreads();
    compute((nt - 1) & 1);

    // ---- writeback: VGPR i -> row half*8+i, col = lane%16 ----
    const int row0 = rowBlk + wm * 32 + half * 8;
    const int col0 = colBlk + wn * 32 + l16;
    float b0 = 0.f, b1 = 0.f;
    if (HASBIAS) {
        const float* bp = bias + (long)z * biasStrideZ;
        b0 = bp[col0];
        b1 = bp[col0 + 16];
    }
    #pragma unroll
    for (int i = 0; i < 8; ++i) {
        C[(long)(row0 + i) * Ndim + col0]           = (CT)(acc00[i] + b0);
        C[(long)(row0 + i) * Ndim + col0 + 16]      = (CT)(acc01[i] + b1);
        C[(long)(row0 + 16 + i) * Ndim + col0]      = (CT)(acc10[i] + b0);
        C[(long)(row0 + 16 + i) * Ndim + col0 + 16] = (CT)(acc11[i] + b1);
    }
}

// ---------------------------------------------------------------------------
// f32 -> bf16 elementwise (vectorized: 4 f32 in, 4 bf16 out per thread)
// ---------------------------------------------------------------------------
__global__ __launch_bounds__(256)
void cvt_bf16(const float4* __restrict__ X, uint2* __restrict__ Y)
{
    long i = (long)blockIdx.x * 256 + threadIdx.x;
    float4 v = X[i];
    union { bf16_t h[4]; uint2 u; } o;
    o.h[0] = (bf16_t)v.x; o.h[1] = (bf16_t)v.y;
    o.h[2] = (bf16_t)v.z; o.h[3] = (bf16_t)v.w;
    Y[i] = o.u;
}

// ---------------------------------------------------------------------------
// f32 [R][C] -> bf16 [C][R]  (32x32 LDS tile transpose, coalesced both sides)
// ---------------------------------------------------------------------------
__global__ __launch_bounds__(256)
void cvt_transpose(const float* __restrict__ X, bf16_t* __restrict__ Y,
                   int R, int C)
{
    __shared__ float tile[32][33];
    const int tx = threadIdx.x & 31;
    const int ty = threadIdx.x >> 5;
    const int r0 = blockIdx.y * 32;
    const int c0 = blockIdx.x * 32;
    #pragma unroll
    for (int i = 0; i < 32; i += 8)
        tile[ty + i][tx] = X[(long)(r0 + ty + i) * C + c0 + tx];
    __syncthreads();
    #pragma unroll
    for (int i = 0; i < 32; i += 8)
        Y[(long)(c0 + ty + i) * R + r0 + tx] = (bf16_t)tile[tx][ty + i];
}

// ---------------------------------------------------------------------------
// tn = rmsnorm(tokens) over D, one wave32 per row, bf16 output
// ---------------------------------------------------------------------------
__global__ __launch_bounds__(256)
void rmsnorm_rows(const float* __restrict__ X, const float* __restrict__ W,
                  bf16_t* __restrict__ Y, int rows, int cols)
{
    int gw   = (blockIdx.x * 256 + threadIdx.x) >> 5;
    int lane = threadIdx.x & 31;
    if (gw >= rows) return;
    const float* xr = X + (long)gw * cols;
    float ss = 0.f;
    for (int j = lane; j < cols; j += 32) { float v = xr[j]; ss += v * v; }
    #pragma unroll
    for (int off = 16; off > 0; off >>= 1) ss += __shfl_xor(ss, off, 32);
    float scale = rsqrtf(ss / (float)cols + EPS_);
    bf16_t* yr = Y + (long)gw * cols;
    for (int j = lane; j < cols; j += 32) yr[j] = (bf16_t)(xr[j] * scale * W[j]);
}

// ---------------------------------------------------------------------------
// Per-head rmsnorm of K half of kv (in place). One wave per (row, h).
// ---------------------------------------------------------------------------
__global__ __launch_bounds__(256)
void knorm_kernel(float* __restrict__ KV, const float* __restrict__ W)
{
    int gw   = (blockIdx.x * 256 + threadIdx.x) >> 5;
    int lane = threadIdx.x & 31;
    int h    = gw & (H_ - 1);
    int row  = gw >> 3;
    float* kp = KV + (long)row * (2 * INNER_) + h * DH_;
    float v0 = kp[lane];
    float v1 = kp[lane + 32];
    float ss = v0 * v0 + v1 * v1;
    #pragma unroll
    for (int off = 16; off > 0; off >>= 1) ss += __shfl_xor(ss, off, 32);
    float scale = rsqrtf(ss / (float)DH_ + EPS_);
    kp[lane]      = v0 * scale * W[lane];
    kp[lane + 32] = v1 * scale * W[lane + 32];
}

// ---------------------------------------------------------------------------
// Softmax attention over M=16 + sigmoid gate; one wave per (l,b,n,h).
// ---------------------------------------------------------------------------
__global__ __launch_bounds__(256)
void attn_kernel(const float* __restrict__ Q, const float* __restrict__ KV,
                 const bf16_t* __restrict__ TN, const float* __restrict__ WG,
                 bf16_t* __restrict__ OG)
{
    int gw   = (blockIdx.x * 256 + threadIdx.x) >> 5;
    int lane = threadIdx.x & 31;
    int h    = gw & (H_ - 1);
    int r    = gw >> 3;
    int bn   = r & (B_ * N_ - 1);

    const float* qp = Q + (long)r * INNER_ + h * DH_;
    float q0 = qp[lane];
    float q1 = qp[lane + 32];

    float sim[M_];
    #pragma unroll
    for (int m = 0; m < M_; ++m) {
        const float* kp = KV + (long)(m * (B_ * N_) + bn) * (2 * INNER_) + h * DH_;
        float p = q0 * kp[lane] + q1 * kp[lane + 32];
        #pragma unroll
        for (int off = 16; off > 0; off >>= 1) p += __shfl_xor(p, off, 32);
        sim[m] = p * 0.125f;
    }
    float mx = sim[0];
    #pragma unroll
    for (int m = 1; m < M_; ++m) mx = fmaxf(mx, sim[m]);
    float den = 0.f;
    #pragma unroll
    for (int m = 0; m < M_; ++m) { sim[m] = __expf(sim[m] - mx); den += sim[m]; }
    float inv = 1.f / den;

    float o0 = 0.f, o1 = 0.f;
    #pragma unroll
    for (int m = 0; m < M_; ++m) {
        const float* vp = KV + (long)(m * (B_ * N_) + bn) * (2 * INNER_) + INNER_ + h * DH_;
        float a = sim[m] * inv;
        o0 += a * vp[lane];
        o1 += a * vp[lane + 32];
    }

    const bf16_t* tp = TN + (long)r * D_;
    float g = 0.f;
    for (int j = lane; j < D_; j += 32) g += (float)tp[j] * WG[j * H_ + h];
    #pragma unroll
    for (int off = 16; off > 0; off >>= 1) g += __shfl_xor(g, off, 32);
    g = 1.f / (1.f + __expf(-g));

    bf16_t* op = OG + (long)r * INNER_ + h * DH_;
    op[lane]      = (bf16_t)(o0 * g);
    op[lane + 32] = (bf16_t)(o1 * g);
}

// ---------------------------------------------------------------------------
extern "C" void kernel_launch(void* const* d_in, const int* in_sizes, int n_in,
                              void* d_out, int out_size, void* d_ws, size_t ws_size,
                              hipStream_t stream)
{
    (void)in_sizes; (void)n_in; (void)out_size; (void)ws_size;

    const float* tokens  = (const float*)d_in[0];   // [L,B,N,D]
    const float* w_net   = (const float*)d_in[1];   // [L,D,D]  (out,in) = [n][k]
    const float* b_net   = (const float*)d_in[2];   // [L,D]
    const float* norm_w  = (const float*)d_in[3];   // [D]
    const float* wq      = (const float*)d_in[4];   // [D,INNER]
    const float* wkv     = (const float*)d_in[5];   // [D,2*INNER]
    const float* knorm_w = (const float*)d_in[6];   // [DH]
    const float* wg      = (const float*)d_in[7];   // [D,H]
    const float* wout    = (const float*)d_in[8];   // [INNER,D]
    float* out = (float*)d_out;                     // [L,B,N,D]

    const long ROWS = (long)L_ * B_ * N_;           // 8192

    char* ws = (char*)d_ws;
    bf16_t* tok_bf  = (bf16_t*)ws; ws += ROWS * D_ * sizeof(bf16_t);
    bf16_t* wnet_bf = (bf16_t*)ws; ws += (long)L_ * D_ * D_ * sizeof(bf16_t);
    bf16_t* wqT     = (bf16_t*)ws; ws += (long)INNER_ * D_ * sizeof(bf16_t);
    bf16_t* wkvT    = (bf16_t*)ws; ws += (long)2 * INNER_ * D_ * sizeof(bf16_t);
    bf16_t* woutT   = (bf16_t*)ws; ws += (long)D_ * INNER_ * sizeof(bf16_t);
    bf16_t* tn      = (bf16_t*)ws; ws += ROWS * D_ * sizeof(bf16_t);
    bf16_t* onet    = (bf16_t*)ws; ws += ROWS * D_ * sizeof(bf16_t);
    float*  q       = (float*)ws;  ws += ROWS * INNER_ * sizeof(float);
    float*  kv      = (float*)ws;  ws += (long)M_ * B_ * N_ * 2 * INNER_ * sizeof(float);
    bf16_t* og      = (bf16_t*)ws; ws += ROWS * INNER_ * sizeof(bf16_t);

    // 0) one-time conversions
    cvt_bf16<<<(int)(ROWS * D_ / 1024), 256, 0, stream>>>(
        (const float4*)tokens, (uint2*)tok_bf);
    cvt_bf16<<<(int)((long)L_ * D_ * D_ / 1024), 256, 0, stream>>>(
        (const float4*)w_net, (uint2*)wnet_bf);
    { dim3 g(INNER_ / 32, D_ / 32);        cvt_transpose<<<g, 256, 0, stream>>>(wq,   wqT,   D_, INNER_); }
    { dim3 g((2 * INNER_) / 32, D_ / 32);  cvt_transpose<<<g, 256, 0, stream>>>(wkv,  wkvT,  D_, 2 * INNER_); }
    { dim3 g(D_ / 32, INNER_ / 32);        cvt_transpose<<<g, 256, 0, stream>>>(wout, woutT, INNER_, D_); }

    // 1) tn = rmsnorm(tokens)
    rmsnorm_rows<<<(int)(ROWS * 32 / 256), 256, 0, stream>>>(tokens, norm_w, tn, (int)ROWS, D_);

    // 2) onet[l] = tokens[l] @ w_net[l]^T + b_net[l]
    {
        dim3 grid(D_ / 64, (B_ * N_) / 128, L_);
        gemm_wmma<true, bf16_t><<<grid, 256, 0, stream>>>(
            tok_bf, wnet_bf, b_net, onet, D_, D_,
            (long)B_ * N_ * D_, (long)D_ * D_, (long)D_, (long)B_ * N_ * D_);
    }
    // 3) q = tn @ wq
    {
        dim3 grid(INNER_ / 64, (int)(ROWS / 128), 1);
        gemm_wmma<false, float><<<grid, 256, 0, stream>>>(
            tn, wqT, nullptr, q, INNER_, D_, 0, 0, 0, 0);
    }
    // 4) kv[0:8] = tokens @ wkv ; kv[8:16] = onet @ wkv
    {
        dim3 grid((2 * INNER_) / 64, (int)(ROWS / 128), 1);
        gemm_wmma<false, float><<<grid, 256, 0, stream>>>(
            tok_bf, wkvT, nullptr, kv, 2 * INNER_, D_, 0, 0, 0, 0);
        gemm_wmma<false, float><<<grid, 256, 0, stream>>>(
            onet, wkvT, nullptr, kv + ROWS * 2 * INNER_, 2 * INNER_, D_, 0, 0, 0, 0);
    }
    // 5) per-head rmsnorm of K half
    {
        long waves = (long)M_ * B_ * N_ * H_;
        knorm_kernel<<<(int)(waves * 32 / 256), 256, 0, stream>>>(kv, knorm_w);
    }
    // 6) attention + gate -> og
    {
        long waves = ROWS * H_;
        attn_kernel<<<(int)(waves * 32 / 256), 256, 0, stream>>>(q, kv, tn, wg, og);
    }
    // 7) pooled = og @ wout -> d_out
    {
        dim3 grid(D_ / 64, (int)(ROWS / 128), 1);
        gemm_wmma<false, float><<<grid, 256, 0, stream>>>(
            og, woutT, nullptr, out, D_, INNER_, 0, 0, 0, 0);
    }
}
